// RnnEncoder_86921548136548
// MI455X (gfx1250) — matrix operand order repaired and checked by reading, not compiled, not run on hardware
//
#include <hip/hip_runtime.h>
#include <cstddef>

// ---------------------------------------------------------------------------
// LSTM  (B=256, T=512, F=128, U=512) for gfx1250 (MI455X)
// Persistent kernel, one device-wide barrier per timestep, bf16 WMMA GEMMs,
// async global->LDS staging (ASYNCcnt), register-resident recurrent weights.
// ---------------------------------------------------------------------------

typedef __attribute__((ext_vector_type(16))) __bf16        bf16x16;
typedef __attribute__((ext_vector_type(8)))  float         f32x8;
typedef __attribute__((ext_vector_type(4)))  unsigned int  u32x4;

#define B_      256
#define T_      512
#define F_      128
#define U_      512
#define G4_     2048          // 4*U
#define KTOT    640           // F + U
#define NKC     20            // KTOT / 32 K-chunks
#define MT      32            // batch-rows per workgroup
#define UT      32            // units per workgroup
#define NWG     128           // (B/MT) * (U/UT) = 8 * 16
#define THREADS 128           // 4 waves (wave32): one wave per gate
#define APAD    8             // LDS row pad (elements) to break bank conflicts
#define AROWS   (KTOT + APAD) // 648 ushorts = 1296 B row stride (16B multiple)

static constexpr size_t WP_ELEMS  = (size_t)G4_ * KTOT;       // packed weights (bf16)
static constexpr size_t WP_BYTES  = WP_ELEMS * 2;             // 2,621,440 B
static constexpr size_t XB_ELEMS  = (size_t)B_ * T_ * F_;     // x in bf16
static constexpr size_t XB_BYTES  = XB_ELEMS * 2;             // 33,554,432 B
static constexpr size_t HBB_ELEMS = 2ull * B_ * U_;           // ping-pong h (bf16)
static constexpr size_t HBB_BYTES = HBB_ELEMS * 2;            // 524,288 B
static constexpr size_t OUT_SEQ   = (size_t)B_ * T_ * U_;     // output sequence elems

union Frag { bf16x16 v; u32x4 q[2]; };

__device__ __forceinline__ unsigned short f2bf(float f) {
  union { float f; unsigned u; } c; c.f = f;
  unsigned u = c.u + 0x7FFFu + ((c.u >> 16) & 1u);   // round-to-nearest-even
  return (unsigned short)(u >> 16);
}

// single v_exp_f32 + v_rcp_f32 each
__device__ __forceinline__ float sigmoid_fast(float x) {
  return __builtin_amdgcn_rcpf(1.0f + __expf(-x));
}
__device__ __forceinline__ float tanh_fast(float x) {
  return 1.0f - 2.0f * __builtin_amdgcn_rcpf(__expf(2.0f * x) + 1.0f);
}

// ---------------------------------------------------------------------------
// Init: pack [W;R] -> column-major bf16 Wp[col][k], convert x -> bf16,
// zero bf16 ping-pong h buffers, reset the grid-barrier counter.
// ---------------------------------------------------------------------------
__global__ void lstm_pack_kernel(const float* __restrict__ x,
                                 const float* __restrict__ W,
                                 const float* __restrict__ R,
                                 unsigned short* __restrict__ Wp,
                                 unsigned short* __restrict__ xbf,
                                 unsigned short* __restrict__ hbf,
                                 unsigned* __restrict__ cnt) {
  size_t idx = (size_t)blockIdx.x * blockDim.x + threadIdx.x;
  if (idx < XB_ELEMS) xbf[idx] = f2bf(x[idx]);            // layout-preserving
  if (idx < WP_ELEMS) {
    int col = (int)(idx / KTOT);        // z column: gate*512 + unit
    int k   = (int)(idx % KTOT);        // 0..127 -> W rows, 128..639 -> R rows
    float v = (k < F_) ? W[(size_t)k * G4_ + col]
                       : R[(size_t)(k - F_) * G4_ + col];
    Wp[idx] = f2bf(v);
  }
  if (idx < HBB_ELEMS) hbf[idx] = 0;                      // bf16 zero
  if (idx == 0)        *cnt = 0u;
}

// ---------------------------------------------------------------------------
// Persistent LSTM kernel. blockIdx.x in [0,128): 8 M-blocks x 16 U-blocks.
// Wave g computes gate g's 32x32 z tile (2x2 WMMA 16x16x32 bf16, K=640).
// B fragments are explicitly preloaded into registers before the time loop.
// ---------------------------------------------------------------------------
__global__ __launch_bounds__(THREADS, 1)
void lstm_persistent_kernel(const unsigned short* __restrict__ xbf,
                            const float* __restrict__ bias,
                            const unsigned short* __restrict__ Wp,
                            unsigned short* __restrict__ hbf,
                            unsigned* __restrict__ cnt,
                            float* __restrict__ out) {
  __shared__ unsigned short sA[MT * AROWS];   // 40.5 KB: [x_t | h] tile, bf16
  __shared__ float          sZ[4][MT * UT];   // 16   KB: gate exchange

  const int tid  = threadIdx.x;
  const int wave = tid >> 5;        // gate index: 0=i 1=f 2=g 3=o
  const int lane = tid & 31;
  const int m0 = (blockIdx.x & 7) * MT;       // batch-row base
  const int u0 = (blockIdx.x >> 3) * UT;      // unit base

  // --- gate-phase mapping: thread owns 8 contiguous units of one row ---
  const int gm  = tid >> 2;          // 0..31 tile row
  const int gu0 = (tid & 3) * 8;     // first unit within tile
  float bI[8], bF[8], bG[8], bO[8];
  float creg[8];
#pragma unroll
  for (int j = 0; j < 8; ++j) {
    int u = u0 + gu0 + j;
    bI[j] = bias[u];
    bF[j] = bias[U_ + u];
    bG[j] = bias[2 * U_ + u];
    bO[j] = bias[3 * U_ + u];
    creg[j] = 0.0f;
  }

  // --- preload ALL B fragments into registers (live across all 512 steps) --
  // B (32x16) as loaded here (contiguous K, matches A): lane<16 -> K 0..15, lane>=16 -> K 16..31, N=lane&15
  Frag fb[NKC][2];                    // 40 fragments = 320 VGPRs of weights
  {
    const unsigned short* bbase[2];
#pragma unroll
    for (int ni = 0; ni < 2; ++ni)
      bbase[ni] = Wp + ((size_t)(wave * U_ + u0 + ni * 16 + (lane & 15))) * KTOT
                     + ((lane >> 4) << 4);
#pragma unroll
    for (int kk = 0; kk < NKC; ++kk)
#pragma unroll
      for (int ni = 0; ni < 2; ++ni) {
        const unsigned short* bp = bbase[ni] + kk * 32;
        fb[kk][ni].q[0] = *(const u32x4*)bp;          // K +0..7  (or +16..23)
        fb[kk][ni].q[1] = *(const u32x4*)(bp + 8);    // K +8..15 (or +24..31)
      }
  }

  // A (16x32) as loaded here (contiguous K, matches B): lane<16 -> K 0..15, lane>=16 -> K 16..31
  const int arow  = lane & 15;
  const int akoff = (lane >> 4) << 4;

  // --- async-staging addressing (per thread: row r, 20 16B chunks) ---------
  const int r_st  = tid & 31;                  // A-tile row this thread stages
  const int c0_st = (tid >> 5) * 8;            // first K of its chunk column
  const int gb_st = m0 + r_st;                 // batch row
  const unsigned lds_row_base =
      (unsigned)(unsigned long long)(const void*)&sA[r_st * AROWS];

  for (int t = 0; t < T_; ++t) {
    const unsigned short* hbprev = hbf + (size_t)(t & 1) * (B_ * U_);
    unsigned short*       hbnext = hbf + (size_t)((t & 1) ^ 1) * (B_ * U_);

    __syncthreads();   // previous step's sA/sZ consumers are done

    // ---- stage A tile: 32 rows x 640 K, pure bf16 async copies -----------
    {
      const unsigned short* xrow = xbf + ((size_t)gb_st * T_ + t) * F_;
      const unsigned short* hrow = hbprev + (size_t)gb_st * U_;
#pragma unroll
      for (int jj = 0; jj < NKC; ++jj) {
        const int k = c0_st + jj * 32;
        const unsigned short* src = (k < F_) ? (xrow + k) : (hrow + (k - F_));
        const unsigned laddr = lds_row_base + (unsigned)(k * 2);
        asm volatile("global_load_async_to_lds_b128 %0, %1, off"
                     :: "v"(laddr), "v"(src) : "memory");
      }
    }
    asm volatile("s_wait_asynccnt 0x0" ::: "memory");
    __syncthreads();

    // ---- GEMM: z[g] = A(32x640) x B_g(640x32), bf16 WMMA f32-acc ---------
    f32x8 acc[2][2];
#pragma unroll
    for (int mi = 0; mi < 2; ++mi)
#pragma unroll
      for (int ni = 0; ni < 2; ++ni)
#pragma unroll
        for (int r = 0; r < 8; ++r) acc[mi][ni][r] = 0.0f;

#pragma unroll
    for (int kk = 0; kk < NKC; ++kk) {
      const int kc = kk * 32;
      Frag fa[2];
#pragma unroll
      for (int mi = 0; mi < 2; ++mi) {
        const unsigned short* ap = &sA[(mi * 16 + arow) * AROWS + kc + akoff];
        fa[mi].q[0] = *(const u32x4*)ap;          // K +0..7  (or +16..23)
        fa[mi].q[1] = *(const u32x4*)(ap + 8);    // K +8..15 (or +24..31)
      }
#pragma unroll
      for (int mi = 0; mi < 2; ++mi)
#pragma unroll
        for (int ni = 0; ni < 2; ++ni)
          acc[mi][ni] = __builtin_amdgcn_wmma_f32_16x16x32_bf16(
              false, fa[mi].v, false, fb[kk][ni].v,
              (short)0, acc[mi][ni], false, false);
    }

    // ---- spill z tiles for cross-gate exchange ---------------------------
    {
      const int zr = (lane >> 4) * 8;   // C/D layout: lane>=16 holds M+8
      const int zc = lane & 15;
#pragma unroll
      for (int mi = 0; mi < 2; ++mi)
#pragma unroll
        for (int ni = 0; ni < 2; ++ni)
#pragma unroll
          for (int r = 0; r < 8; ++r)
            sZ[wave][(mi * 16 + zr + r) * UT + ni * 16 + zc] = acc[mi][ni][r];
    }
    __syncthreads();

    // ---- gate nonlinearity, cell update (c in registers) -----------------
    {
      union { unsigned short h[8]; u32x4 q; } hv;
      const int gb = m0 + gm;
#pragma unroll
      for (int j = 0; j < 8; ++j) {
        int idx = gm * UT + gu0 + j;
        float zi = sZ[0][idx] + bI[j];
        float zf = sZ[1][idx] + bF[j];
        float zg = sZ[2][idx] + bG[j];
        float zo = sZ[3][idx] + bO[j];
        float ig = sigmoid_fast(zi);
        float fg = sigmoid_fast(zf);
        float gg = tanh_fast(zg);
        float og = sigmoid_fast(zo);
        float c  = fg * creg[j] + ig * gg;
        creg[j]  = c;
        float h  = og * tanh_fast(c);
        hv.h[j]  = f2bf(h);
        int gu = u0 + gu0 + j;
        out[((size_t)gb * T_ + t) * U_ + gu] = h;
        if (t == T_ - 1) {
          out[OUT_SEQ + (size_t)gb * U_ + gu] = h;                    // h_last
          out[OUT_SEQ + (size_t)B_ * U_ + (size_t)gb * U_ + gu] = c;  // c_last
        }
      }
      // one coalesced 16B bf16 store of this thread's h slice
      *(u32x4*)&hbnext[(size_t)gb * U_ + u0 + gu0] = hv.q;
    }

    // ---- device-wide barrier (h ping-pong => one barrier per step) -------
    __threadfence();     // make this thread's h stores agent-visible
    __syncthreads();
    if (tid == 0) {
      __hip_atomic_fetch_add(cnt, 1u, __ATOMIC_ACQ_REL, __HIP_MEMORY_SCOPE_AGENT);
      const unsigned target = (unsigned)NWG * (unsigned)(t + 1);
      while (__hip_atomic_load(cnt, __ATOMIC_ACQUIRE, __HIP_MEMORY_SCOPE_AGENT) < target) {
        __builtin_amdgcn_s_sleep(4);
      }
    }
    __syncthreads();
    __threadfence();     // acquire: drop stale near-cache lines before reading h
  }
}

// ---------------------------------------------------------------------------
extern "C" void kernel_launch(void* const* d_in, const int* in_sizes, int n_in,
                              void* d_out, int out_size, void* d_ws, size_t ws_size,
                              hipStream_t stream) {
  const float* x    = (const float*)d_in[0];   // [B,T,F]
  const float* W    = (const float*)d_in[1];   // [F,4U]
  const float* R    = (const float*)d_in[2];   // [U,4U]
  const float* bias = (const float*)d_in[3];   // [4U]
  float* out        = (float*)d_out;           // [B,T,U] ++ h_last ++ c_last

  unsigned short* Wp  = (unsigned short*)d_ws;
  unsigned short* xbf = (unsigned short*)((char*)d_ws + WP_BYTES);
  unsigned short* hbf = (unsigned short*)((char*)d_ws + WP_BYTES + XB_BYTES);
  unsigned*       cnt = (unsigned*)((char*)d_ws + WP_BYTES + XB_BYTES + HBB_BYTES);

  const int packBlocks = (int)((XB_ELEMS + 255) / 256);   // covers all init work
  lstm_pack_kernel<<<packBlocks, 256, 0, stream>>>(x, W, R, Wp, xbf, hbf, cnt);
  lstm_persistent_kernel<<<NWG, THREADS, 0, stream>>>(xbf, bias, Wp, hbf, cnt, out);
}
